// MultiheadAttention_81793357185843
// MI455X (gfx1250) — matrix operand order repaired and checked by reading, not compiled, not run on hardware
//
#include <hip/hip_runtime.h>
#include <hip/hip_bf16.h>
#include <stdint.h>

typedef _Float16 h16v __attribute__((ext_vector_type(16)));
typedef _Float16 h8v  __attribute__((ext_vector_type(8)));
typedef _Float16 h4v  __attribute__((ext_vector_type(4)));
typedef float    f8v  __attribute__((ext_vector_type(8)));
typedef float    f4v  __attribute__((ext_vector_type(4)));

#define NN  512          // nodes
#define NG  16           // graphs
#define EMB 768
#define NH  12
#define HD  64
#define NB  (NG * NH)    // 192 attention batches
#define NT  (NN * NG)    // 8192 tokens
#define NKB (EMB / 32)   // 24 K-steps for the projections

__device__ __forceinline__ float redmax16(float v) {
#pragma unroll
    for (int o = 1; o < 16; o <<= 1) v = fmaxf(v, __shfl_xor(v, o, 32));
    return v;
}
__device__ __forceinline__ float redsum16(float v) {
#pragma unroll
    for (int o = 1; o < 16; o <<= 1) v += __shfl_xor(v, o, 32);
    return v;
}
__device__ __forceinline__ f8v zero8() {
    f8v z;
#pragma unroll
    for (int i = 0; i < 8; i++) z[i] = 0.0f;
    return z;
}
__device__ __forceinline__ h16v cat16(h8v lo, h8v hi) {
    h16v r;
#pragma unroll
    for (int i = 0; i < 8; i++) { r[i] = lo[i]; r[8 + i] = hi[i]; }
    return r;
}

// ---------------------------------------------------------------------------
// Kernel 1: fused QKV projection.  Y = (X*W + b) [* scale for Q]
// Double-buffered LDS, register-staged loads via running pointers.
// ---------------------------------------------------------------------------
__global__ __launch_bounds__(128) void mha_qkv_proj(
    const float* __restrict__ x,
    const float* __restrict__ Wq, const float* __restrict__ Wk, const float* __restrict__ Wv,
    const float* __restrict__ bq, const float* __restrict__ bk, const float* __restrict__ bv,
    _Float16* __restrict__ Qh, _Float16* __restrict__ Kh, _Float16* __restrict__ Vh)
{
    const int z = blockIdx.z;
    const float* W    = (z == 0) ? Wq : (z == 1) ? Wk : Wv;
    const float* bias = (z == 0) ? bq : (z == 1) ? bk : bv;
    _Float16*    dst  = (z == 0) ? Qh : (z == 1) ? Kh : Vh;
    const float scale = (z == 0) ? 0.125f : 1.0f;   // HEAD_DIM^-0.5

    __shared__ _Float16 Xs[2][64][40];    // 64 rows x 32 k (stride 40 halfs = 80B)
    __shared__ _Float16 Wts[2][64][40];   // transposed: [out col][k]

    const int tid  = threadIdx.x;
    const int lane = tid & 31, wv = tid >> 5;
    const int lh = lane >> 4, ln = lane & 15;
    const int tbase = blockIdx.y * 64;
    const int nbase = blockIdx.x * 64;

    // running pointers: per-it deltas are compile-time, fold into IOFFSET
    const float* xRun = x + (size_t)(tbase + (tid >> 3)) * EMB + (tid & 7) * 4;
    const float* wRun = W + (size_t)(tid >> 4) * EMB + nbase + (tid & 15) * 4;
    _Float16* xsB  = &Xs[0][tid >> 3][(tid & 7) * 4];
    _Float16* wtsB = &Wts[0][(tid & 15) * 4][tid >> 4];

    f4v xr[4], wr[4];
    auto ldTile = [&]() {
#pragma unroll
        for (int it = 0; it < 4; it++) xr[it] = *(const f4v*)(xRun + (size_t)it * 16 * EMB);
#pragma unroll
        for (int it = 0; it < 4; it++) wr[it] = *(const f4v*)(wRun + (size_t)it * 8 * EMB);
        xRun += 32;
        wRun += (size_t)32 * EMB;
    };
    auto stTile = [&](int buf) {
        const int bo = buf * 64 * 40;   // halfs per buffer
#pragma unroll
        for (int it = 0; it < 4; it++) {
            h4v xh;
#pragma unroll
            for (int i = 0; i < 4; i++) xh[i] = (_Float16)xr[it][i];
            *(h4v*)(xsB + bo + it * 16 * 40) = xh;             // ds_store_b64
#pragma unroll
            for (int i = 0; i < 4; i++)
                wtsB[bo + it * 8 + i * 40] = (_Float16)wr[it][i];
        }
    };

    f8v acc[4];
#pragma unroll
    for (int t = 0; t < 4; t++) acc[t] = zero8();

    ldTile();
    stTile(0);
    for (int kk = 0; kk < NKB; kk++) {
        const int buf = kk & 1;
        __syncthreads();
        if (kk + 1 < NKB) ldTile();                  // next tile in flight

        const _Float16* ap = &Xs[buf][wv * 16 + ln][0];
        h16v a = cat16(*(const h8v*)(ap + lh * 8), *(const h8v*)(ap + 16 + lh * 8));
        h16v b[4];
#pragma unroll
        for (int t = 0; t < 4; t++) {
            const _Float16* bp = &Wts[buf][t * 16 + ln][0];
            b[t] = cat16(*(const h8v*)(bp + lh * 8), *(const h8v*)(bp + 16 + lh * 8));
        }
#pragma unroll
        for (int t = 0; t < 4; t++)
            acc[t] = __builtin_amdgcn_wmma_f32_16x16x32_f16(
                false, a, false, b[t], (short)0, acc[t], false, false);

        if (kk + 1 < NKB) stTile((kk + 1) & 1);      // fill other buffer
    }

    // epilogue: +bias, *scale, scatter into head-major layout
#pragma unroll
    for (int t = 0; t < 4; t++) {
        int colg = nbase + t * 16 + ln;
        int h = colg >> 6, d = colg & 63;
        float bval = bias[colg];
#pragma unroll
        for (int r = 0; r < 8; r++) {
            int token = tbase + wv * 16 + lh * 8 + r;
            int node = token >> 4, g = token & 15;
            float y = (acc[t][r] + bval) * scale;
            dst[((size_t)(g * NH + h) * NN + node) * HD + d] = (_Float16)y;
        }
    }
}

// ---------------------------------------------------------------------------
// Kernel 2: flash attention per batch. Block = 4 waves, 64 rows; wave = 16 rows.
// Double-buffered K + transposed-V LDS chunks (K deduped across waves).
// ---------------------------------------------------------------------------
__global__ __launch_bounds__(128) void mha_attn(
    const _Float16* __restrict__ Qh, const _Float16* __restrict__ Kh,
    const _Float16* __restrict__ Vh,
    const float* __restrict__ attn_bias, const float* __restrict__ attn_mask,
    const uint8_t* __restrict__ kpm, _Float16* __restrict__ aws)
{
    __shared__ _Float16 Vt[2][64][40];    // V^T chunk: [d][32 keys]
    __shared__ _Float16 Ks[2][32][72];    // K chunk: [key][64 d], stride 72 halfs
    __shared__ _Float16 Pst[4][16][32];   // per-wave P staging (C->A relayout)

    const int tid  = threadIdx.x;
    const int lane = tid & 31, wv = tid >> 5;
    const int lh = lane >> 4, ln = lane & 15;
    const int batch = blockIdx.y;
    const int g = batch / NH, h = batch % NH;
    const int r0 = blockIdx.x * 64 + wv * 16;

    const _Float16* Qb = Qh + (size_t)batch * NN * HD;

    // chunk staging: thread owns row tid>>2 (of 32 keys), 16 halfs at (tid&3)*16
    const int cn = tid >> 2, cd = (tid & 3) * 16;
    const _Float16* kRun = Kh + (size_t)batch * NN * HD + (size_t)cn * HD + cd;
    const _Float16* vRun = Vh + (size_t)batch * NN * HD + (size_t)cn * HD + cd;
    h8v kr0, kr1, vr0, vr1;
    auto ldChunk = [&]() {
        kr0 = *(const h8v*)(kRun);
        kr1 = *(const h8v*)(kRun + 8);
        vr0 = *(const h8v*)(vRun);
        vr1 = *(const h8v*)(vRun + 8);
        kRun += 32 * HD;
        vRun += 32 * HD;
    };
    auto stChunk = [&](int buf) {
        *(h8v*)&Ks[buf][cn][cd]     = kr0;            // ds_store_b128
        *(h8v*)&Ks[buf][cn][cd + 8] = kr1;
#pragma unroll
        for (int i = 0; i < 8; i++) {                 // V transposed scatter
            Vt[buf][cd + i][cn]     = vr0[i];
            Vt[buf][cd + 8 + i][cn] = vr1[i];
        }
    };

    // Q A-fragments for this wave's 16 rows (K = 0..31 and 32..63)
    h16v a0, a1;
    {
        const _Float16* qp = Qb + (size_t)(r0 + ln) * HD;
        a0 = cat16(*(const h8v*)(qp + lh * 8), *(const h8v*)(qp + 16 + lh * 8));
        a1 = cat16(*(const h8v*)(qp + 32 + lh * 8), *(const h8v*)(qp + 48 + lh * 8));
    }

    // running pointers for the bias/mask/padding stream (step 32 cols per j)
    const float* biasRun = attn_bias + ((size_t)batch * NN + r0 + lh * 8) * NN + ln;
    const float* maskRun = attn_mask + (size_t)(r0 + lh * 8) * NN + ln;
    const uint8_t* kpmRun = kpm + g * NN + ln;

    f8v acc[4];
#pragma unroll
    for (int t = 0; t < 4; t++) acc[t] = zero8();
    float M[8], L[8];
#pragma unroll
    for (int r = 0; r < 8; r++) { M[r] = -1e30f; L[r] = 0.0f; }

    ldChunk();
    stChunk(0);
    for (int j = 0; j < 16; j++) {
        const int buf = j & 1;
        __syncthreads();
        if (j < 15) {
            ldChunk();                                 // next K/V chunk in flight
            __builtin_prefetch(biasRun + 32, 0, 1);    // next bias columns
        }

        // hoisted: bias/mask/padding for both 16-col tiles (const offsets fold)
        float biasv[2][8];
        bool  pmv[2];
#pragma unroll
        for (int t = 0; t < 2; t++) {
            pmv[t] = kpmRun[t * 16] != 0;
#pragma unroll
            for (int r = 0; r < 8; r++)
                biasv[t][r] = biasRun[(size_t)r * NN + t * 16]
                            + maskRun[(size_t)r * NN + t * 16];
        }
        biasRun += 32; maskRun += 32; kpmRun += 32;

        // K B-frags from LDS chunk (k contiguous, 72-half stride spreads banks)
        h16v kb0[2], kb1[2];
#pragma unroll
        for (int t = 0; t < 2; t++) {
            const _Float16* kp = &Ks[buf][t * 16 + ln][0];
            kb0[t] = cat16(*(const h8v*)(kp + lh * 8), *(const h8v*)(kp + 16 + lh * 8));
            kb1[t] = cat16(*(const h8v*)(kp + 32 + lh * 8), *(const h8v*)(kp + 48 + lh * 8));
        }

        // two 16x16 score tiles
        f8v s[2];
#pragma unroll
        for (int t = 0; t < 2; t++) {
            f8v c = zero8();
            c = __builtin_amdgcn_wmma_f32_16x16x32_f16(false, a0, false, kb0[t], (short)0, c, false, false);
            c = __builtin_amdgcn_wmma_f32_16x16x32_f16(false, a1, false, kb1[t], (short)0, c, false, false);
#pragma unroll
            for (int r = 0; r < 8; r++)
                s[t][r] = pmv[t] ? -1e30f : (c[r] + biasv[t][r]);
        }

        // online softmax update + stage P (f16) for A-frag relayout
        float p0[8], p1[8];
#pragma unroll
        for (int r = 0; r < 8; r++) {
            float tm   = redmax16(fmaxf(s[0][r], s[1][r]));
            float nm   = fmaxf(M[r], tm);
            float corr = __expf(M[r] - nm);
            M[r] = nm;
            p0[r] = __expf(s[0][r] - nm);
            p1[r] = __expf(s[1][r] - nm);
            float rs = redsum16(p0[r] + p1[r]);
            L[r] = L[r] * corr + rs;
#pragma unroll
            for (int t = 0; t < 4; t++) acc[t][r] *= corr;
            Pst[wv][lh * 8 + r][ln]      = (_Float16)p0[r];
            Pst[wv][lh * 8 + r][16 + ln] = (_Float16)p1[r];
        }

        // read back P as 16x32 A-fragment (same-wave LDS RAW, DScnt ordered)
        h16v ap;
        {
            const _Float16* pp = &Pst[wv][ln][0];
            ap = cat16(*(const h8v*)(pp + lh * 8), *(const h8v*)(pp + 16 + lh * 8));
        }
        // acc += P * V  (V B-frags from transposed LDS chunk)
        h16v bv16[4];
#pragma unroll
        for (int t = 0; t < 4; t++) {
            const _Float16* vp = &Vt[buf][t * 16 + ln][0];
            bv16[t] = cat16(*(const h8v*)(vp + lh * 8), *(const h8v*)(vp + 16 + lh * 8));
        }
#pragma unroll
        for (int t = 0; t < 4; t++)
            acc[t] = __builtin_amdgcn_wmma_f32_16x16x32_f16(
                false, ap, false, bv16[t], (short)0, acc[t], false, false);

        if (j < 15) stChunk((j + 1) & 1);              // deposit next chunk
    }

    // finalize: divide by row sum, scatter to token-major [token][E] f16
#pragma unroll
    for (int t = 0; t < 4; t++) {
#pragma unroll
        for (int r = 0; r < 8; r++) {
            int row = r0 + lh * 8 + r;
            float o = acc[t][r] / L[r];
            int token = row * NG + g;
            aws[(size_t)token * EMB + h * HD + t * 16 + ln] = (_Float16)o;
        }
    }
}

// ---------------------------------------------------------------------------
// Kernel 3: output projection.  out = attn(f16) * Wo + bo, f32 out.
// ---------------------------------------------------------------------------
__global__ __launch_bounds__(128) void mha_out_proj(
    const _Float16* __restrict__ aws, const float* __restrict__ Wo,
    const float* __restrict__ bo, float* __restrict__ out)
{
    __shared__ _Float16 Xs[2][64][40];
    __shared__ _Float16 Wts[2][64][40];

    const int tid  = threadIdx.x;
    const int lane = tid & 31, wv = tid >> 5;
    const int lh = lane >> 4, ln = lane & 15;
    const int tbase = blockIdx.y * 64;
    const int nbase = blockIdx.x * 64;

    // running pointers
    const _Float16* xRun = aws + (size_t)(tbase + (tid >> 1)) * EMB + (tid & 1) * 16;
    const float*    wRun = Wo + (size_t)(tid >> 4) * EMB + nbase + (tid & 15) * 4;
    _Float16* xsB  = &Xs[0][tid >> 1][(tid & 1) * 16];
    _Float16* wtsB = &Wts[0][(tid & 15) * 4][tid >> 4];

    h8v xr0, xr1;
    f4v wr[4];
    auto ldTile = [&]() {
        xr0 = *(const h8v*)(xRun);
        xr1 = *(const h8v*)(xRun + 8);
#pragma unroll
        for (int it = 0; it < 4; it++) wr[it] = *(const f4v*)(wRun + (size_t)it * 8 * EMB);
        xRun += 32;
        wRun += (size_t)32 * EMB;
    };
    auto stTile = [&](int buf) {
        const int bo = buf * 64 * 40;
        *(h8v*)(xsB + bo)     = xr0;     // ds_store_b128
        *(h8v*)(xsB + bo + 8) = xr1;
#pragma unroll
        for (int it = 0; it < 4; it++)
#pragma unroll
            for (int i = 0; i < 4; i++)
                wtsB[bo + it * 8 + i * 40] = (_Float16)wr[it][i];
    };

    f8v acc[4];
#pragma unroll
    for (int t = 0; t < 4; t++) acc[t] = zero8();

    ldTile();
    stTile(0);
    for (int kk = 0; kk < NKB; kk++) {
        const int buf = kk & 1;
        __syncthreads();
        if (kk + 1 < NKB) ldTile();

        const _Float16* ap = &Xs[buf][wv * 16 + ln][0];
        h16v a = cat16(*(const h8v*)(ap + lh * 8), *(const h8v*)(ap + 16 + lh * 8));
        h16v b[4];
#pragma unroll
        for (int t = 0; t < 4; t++) {
            const _Float16* bp = &Wts[buf][t * 16 + ln][0];
            b[t] = cat16(*(const h8v*)(bp + lh * 8), *(const h8v*)(bp + 16 + lh * 8));
        }
#pragma unroll
        for (int t = 0; t < 4; t++)
            acc[t] = __builtin_amdgcn_wmma_f32_16x16x32_f16(
                false, a, false, b[t], (short)0, acc[t], false, false);

        if (kk + 1 < NKB) stTile((kk + 1) & 1);
    }

#pragma unroll
    for (int t = 0; t < 4; t++) {
        int colg = nbase + t * 16 + ln;
        float bval = bo[colg];
#pragma unroll
        for (int r = 0; r < 8; r++) {
            int token = tbase + wv * 16 + lh * 8 + r;
            out[(size_t)token * EMB + colg] = acc[t][r] + bval;
        }
    }
}

// ---------------------------------------------------------------------------
extern "C" void kernel_launch(void* const* d_in, const int* in_sizes, int n_in,
                              void* d_out, int out_size, void* d_ws, size_t ws_size,
                              hipStream_t stream) {
    const float*   x         = (const float*)d_in[0];
    const float*   attn_bias = (const float*)d_in[1];
    const uint8_t* kpm       = (const uint8_t*)d_in[2];
    const float*   attn_mask = (const float*)d_in[3];
    const float*   Wq = (const float*)d_in[4];  const float* bq = (const float*)d_in[5];
    const float*   Wk = (const float*)d_in[6];  const float* bk = (const float*)d_in[7];
    const float*   Wv = (const float*)d_in[8];  const float* bv = (const float*)d_in[9];
    const float*   Wo = (const float*)d_in[10]; const float* bo = (const float*)d_in[11];
    float* out = (float*)d_out;

    const size_t headElems = (size_t)NB * NN * HD;      // 6,291,456 halfs each
    _Float16* Qh  = (_Float16*)d_ws;
    _Float16* Kh  = Qh + headElems;
    _Float16* Vh  = Kh + headElems;
    _Float16* aws = Vh + headElems;                     // [8192][768] f16

    mha_qkv_proj<<<dim3(EMB / 64, NT / 64, 3), 128, 0, stream>>>(
        x, Wq, Wk, Wv, bq, bk, bv, Qh, Kh, Vh);
    mha_attn<<<dim3(NN / 64, NB), 128, 0, stream>>>(
        Qh, Kh, Vh, attn_bias, attn_mask, kpm, aws);
    mha_out_proj<<<dim3(EMB / 64, NT / 64), 128, 0, stream>>>(
        aws, Wo, bo, out);
}
